// Attention_43104291783076
// MI455X (gfx1250) — matrix operand order repaired and checked by reading, not compiled, not run on hardware
//
#include <hip/hip_runtime.h>
#include <hip/hip_bf16.h>

typedef __bf16 bf16_t;
typedef __attribute__((ext_vector_type(8)))  __bf16 v8bf;
typedef __attribute__((ext_vector_type(16))) __bf16 v16bf;
typedef __attribute__((ext_vector_type(8)))  float  v8f;
typedef unsigned int v4u __attribute__((ext_vector_type(4)));
typedef int          v8i __attribute__((ext_vector_type(8)));
typedef int          v4i __attribute__((ext_vector_type(4)));

#define C_DIM   1024
#define NHEAD   16
#define HDIM    64
#define SEQ     2048
#define BATCH   2
#define MTOT    (BATCH * SEQ)   /* 4096 */
#define NQKV    (3 * C_DIM)     /* 3072 */
#define SCALE_F 0.125f          /* 64^-0.5 */

#if __has_builtin(__builtin_amdgcn_sched_group_barrier)
#define SCHED_GROUP(mask, size, id) __builtin_amdgcn_sched_group_barrier(mask, size, id)
#else
#define SCHED_GROUP(mask, size, id)
#endif

// ---------------------------------------------------------------------------
// WMMA fragment helpers (bf16 16x16x32).
// A 16x32: lane l holds row (l&15); elems 0..7 at K=(l>>4)*8, 8..15 at +16.
// B 32x16: lane l holds col (l&15); elems 0..7 at K=(l>>4)*8, 8..15 at +16.
// C/D:     VGPR r, lane l -> row r+8*(l>>4), col (l&15).
// ---------------------------------------------------------------------------
static __device__ __forceinline__ v16bf load_a_frag(const bf16_t* p) {
  v8bf lo = *(const v8bf*)p;
  v8bf hi = *(const v8bf*)(p + 16);
  v16bf f;
#pragma unroll
  for (int i = 0; i < 8; ++i) { f[i] = lo[i]; f[i + 8] = hi[i]; }
  return f;
}
static __device__ __forceinline__ v16bf load_b_frag(const bf16_t* p) {
  v8bf lo = *(const v8bf*)p;
  v8bf hi = *(const v8bf*)(p + 16);
  v16bf f;
#pragma unroll
  for (int i = 0; i < 8; ++i) { f[i] = lo[i]; f[i + 8] = hi[i]; }
  return f;
}
static __device__ __forceinline__ v8f wmma_bf16(v16bf a, v16bf b, v8f c) {
  return __builtin_amdgcn_wmma_f32_16x16x32_bf16(false, a, false, b,
                                                 (short)0, c, false, false);
}
static __device__ __forceinline__ float half_max(float v) {
  v = fmaxf(v, __shfl_xor(v, 1, 32));
  v = fmaxf(v, __shfl_xor(v, 2, 32));
  v = fmaxf(v, __shfl_xor(v, 4, 32));
  v = fmaxf(v, __shfl_xor(v, 8, 32));
  return v;
}
static __device__ __forceinline__ float half_sum(float v) {
  v += __shfl_xor(v, 1, 32);
  v += __shfl_xor(v, 2, 32);
  v += __shfl_xor(v, 4, 32);
  v += __shfl_xor(v, 8, 32);
  return v;
}

// ---------------------------------------------------------------------------
// Tensor Data Mover: async 2D bf16 tile load Global -> LDS (ISA ch.8 D#).
// ---------------------------------------------------------------------------
static __device__ __forceinline__ unsigned lds_off(const void* p) {
  return (unsigned)(unsigned long long)p;   // flat LDS addr truncates to offset
}
static __device__ __forceinline__ void tdm_load_2d_bf16(
    unsigned lds_addr, const bf16_t* gptr,
    unsigned tile_d0, unsigned tile_d1,
    unsigned tensor_d0, unsigned tensor_d1, unsigned stride_d0) {
  unsigned long long ga = (unsigned long long)gptr;
  v4u g0;
  g0[0] = 1u;                                          // count=1 (valid), user mode
  g0[1] = lds_addr;                                    // LDS byte address
  g0[2] = (unsigned)(ga & 0xffffffffu);                // global_addr[31:0]
  g0[3] = (unsigned)((ga >> 32) & 0x1ffffffu) | (2u << 30); // addr[56:32] | type=2
  v8i g1;
  g1[0] = (int)(1u << 16);                             // wg_mask=0, data_size=1 (2B)
  g1[1] = (int)((tensor_d0 & 0xffffu) << 16);          // tensor_dim0[15:0] @bit48
  g1[2] = (int)((tensor_d0 >> 16) | ((tensor_d1 & 0xffffu) << 16));
  g1[3] = (int)((tensor_d1 >> 16) | ((tile_d0 & 0xffffu) << 16));
  g1[4] = (int)(tile_d1 & 0xffffu);                    // tile_dim1 | tile_dim2=0
  g1[5] = (int)stride_d0;                              // tensor_dim0_stride[31:0]
  g1[6] = 0;
  g1[7] = 0;
  v4i z = {0, 0, 0, 0};
#if defined(__clang_major__) && (__clang_major__ >= 23)
  v8i z8 = {0, 0, 0, 0, 0, 0, 0, 0};
  __builtin_amdgcn_tensor_load_to_lds(g0, g1, z, z, z8, 0);
#else
  __builtin_amdgcn_tensor_load_to_lds(g0, g1, z, z, 0);
#endif
}

// ---------------------------------------------------------------------------
// Kernel 1: fp32 -> bf16 conversion
// ---------------------------------------------------------------------------
__global__ void __launch_bounds__(256) cvt_f32_bf16(const float* __restrict__ src,
                                                    bf16_t* __restrict__ dst, int n) {
  int i = blockIdx.x * 256 + threadIdx.x;
  if (i < n) dst[i] = (bf16_t)src[i];
}

// ---------------------------------------------------------------------------
// GEMM inner step: 4 B-frags in distinct registers (8 ds_load_b128 issued
// before the 4 WMMAs, enforced with sched_group_barrier) -> waits pipeline
// as dscnt 6/4/2/0 instead of 0 before every WMMA.
// ---------------------------------------------------------------------------
static __device__ __forceinline__ void gemm_step(v8f acc[4], v16bf a,
                                                 const bf16_t* bt,
                                                 int c, int boff) {
  v16bf b0 = load_b_frag(bt + (0 * 16 + c) * 32 + boff);
  v16bf b1 = load_b_frag(bt + (1 * 16 + c) * 32 + boff);
  v16bf b2 = load_b_frag(bt + (2 * 16 + c) * 32 + boff);
  v16bf b3 = load_b_frag(bt + (3 * 16 + c) * 32 + boff);
  acc[0] = wmma_bf16(a, b0, acc[0]);
  acc[1] = wmma_bf16(a, b1, acc[1]);
  acc[2] = wmma_bf16(a, b2, acc[2]);
  acc[3] = wmma_bf16(a, b3, acc[3]);
  SCHED_GROUP(0x100, 8, 0);   // group: 8 DS reads first
  SCHED_GROUP(0x008, 4, 0);   // then: 4 WMMAs
}

// ---------------------------------------------------------------------------
// Kernel 2: QKV GEMM (M=4096, N=3072, K=1024) + bias + scatter.
// 8 waves, block tile 128x64, wave tile 16x64. B tiles via TDM into
// double-buffered LDS; k-loop modulo-2 pipelined (aA/aB, const buffers).
// ---------------------------------------------------------------------------
__global__ void __launch_bounds__(256) qkv_gemm(const bf16_t* __restrict__ xb,
                                                const bf16_t* __restrict__ wib,
                                                const float* __restrict__ b_in,
                                                bf16_t* __restrict__ qb,
                                                bf16_t* __restrict__ kb,
                                                bf16_t* __restrict__ vtb) {
  __shared__ __align__(16) bf16_t btile[2][64 * 32];

  const int lane = threadIdx.x & 31;
  const int wave = threadIdx.x >> 5;
  const int c    = lane & 15;
  const int aoff = (lane >> 4) * 8;
  const int boff = (lane >> 4) * 8;
  const int row_a = blockIdx.x * 128 + wave * 16 + c;
  const int col0  = blockIdx.y * 64;

  const bf16_t* arow = xb + (size_t)row_a * C_DIM + aoff;
  const bf16_t* bsrc = wib + (size_t)col0 * C_DIM;

  if (wave == 0)
    tdm_load_2d_bf16(lds_off(&btile[0][0]), bsrc, 32, 64, C_DIM, NQKV, C_DIM);

  v8f acc[4] = {};
  v16bf aA = load_a_frag(arow);
  v16bf aB = aA;

  const int NI = C_DIM / 32;                           // 32 k-steps, 16 pairs
  for (int j = 0; j < NI / 2 - 1; ++j) {
    const int i0 = 2 * j;
    // even step: compute buf0, DMA buf1
    if (wave == 0) __builtin_amdgcn_s_wait_tensorcnt((short)0);
    __syncthreads();
    if (wave == 0)
      tdm_load_2d_bf16(lds_off(&btile[1][0]), bsrc + (i0 + 1) * 32,
                       32, 64, C_DIM, NQKV, C_DIM);
    aB = load_a_frag(arow + (i0 + 1) * 32);
    gemm_step(acc, aA, &btile[0][0], c, boff);
    // odd step: compute buf1, DMA buf0
    if (wave == 0) __builtin_amdgcn_s_wait_tensorcnt((short)0);
    __syncthreads();
    if (wave == 0)
      tdm_load_2d_bf16(lds_off(&btile[0][0]), bsrc + (i0 + 2) * 32,
                       32, 64, C_DIM, NQKV, C_DIM);
    aA = load_a_frag(arow + (i0 + 2) * 32);
    gemm_step(acc, aB, &btile[1][0], c, boff);
  }
  // final pair (steps NI-2, NI-1)
  if (wave == 0) __builtin_amdgcn_s_wait_tensorcnt((short)0);
  __syncthreads();
  if (wave == 0)
    tdm_load_2d_bf16(lds_off(&btile[1][0]), bsrc + (NI - 1) * 32,
                     32, 64, C_DIM, NQKV, C_DIM);
  aB = load_a_frag(arow + (NI - 1) * 32);
  gemm_step(acc, aA, &btile[0][0], c, boff);
  if (wave == 0) __builtin_amdgcn_s_wait_tensorcnt((short)0);
  __syncthreads();
  gemm_step(acc, aB, &btile[1][0], c, boff);

#pragma unroll
  for (int t = 0; t < 4; ++t) {
    const int o = col0 + t * 16 + c;
    const float bias = b_in[o];
    const int which = o >> 10;
    const int rem = o & 1023;
    const int h = rem >> 6, d = rem & 63;
#pragma unroll
    for (int r = 0; r < 8; ++r) {
      const int m = blockIdx.x * 128 + wave * 16 + ((lane >> 4) * 8) + r;
      const int b = m >> 11, s = m & 2047;
      const int bh = b * NHEAD + h;
      const float v = acc[t][r] + bias;
      if (which == 0)      qb[((size_t)bh * SEQ + s) * HDIM + d] = (bf16_t)(v * SCALE_F);
      else if (which == 1) kb[((size_t)bh * SEQ + s) * HDIM + d] = (bf16_t)v;
      else                 vtb[((size_t)bh * HDIM + d) * SEQ + s] = (bf16_t)v;
    }
  }
}

// ---------------------------------------------------------------------------
// Kernel 3: flash attention. 4 waves/block, one 16-row query tile per wave,
// all 4 tiles in the same (b,h); K/V 32-key blocks shared via TDM-staged
// double-buffered LDS. Online softmax; P via per-wave LDS round trip.
// ---------------------------------------------------------------------------
__global__ void __launch_bounds__(128) flash_attn(const bf16_t* __restrict__ qb,
                                                  const bf16_t* __restrict__ kb,
                                                  const bf16_t* __restrict__ vtb,
                                                  bf16_t* __restrict__ ob) {
  __shared__ __align__(16) bf16_t ktile[2][32 * 64];   // keys x d
  __shared__ __align__(16) bf16_t vtile[2][64 * 32];   // d x keys
  __shared__ __align__(16) bf16_t pbuf[4][16 * 32];    // per-wave P

  const int lane = threadIdx.x & 31;
  const int wave = threadIdx.x >> 5;
  const int c    = lane & 15;
  const int aoff = (lane >> 4) * 8;
  const int boff = (lane >> 4) * 8;

  const int tile = blockIdx.x * 4 + wave;
  const int bh   = tile >> 7;
  const int row0 = (tile & 127) * 16;

  const bf16_t* qbase = qb  + (size_t)bh * SEQ * HDIM;
  const bf16_t* kbase = kb  + (size_t)bh * SEQ * HDIM;
  const bf16_t* vbase = vtb + (size_t)bh * HDIM * SEQ;

  if (wave == 0) {
    tdm_load_2d_bf16(lds_off(&ktile[0][0]), kbase, HDIM, 32, HDIM, SEQ, HDIM);
    tdm_load_2d_bf16(lds_off(&vtile[0][0]), vbase, 32, HDIM, SEQ, HDIM, SEQ);
  }

  const v16bf qa0 = load_a_frag(qbase + (size_t)(row0 + c) * HDIM + 0  + aoff);
  const v16bf qa1 = load_a_frag(qbase + (size_t)(row0 + c) * HDIM + 32 + aoff);

  float mrow[8], lrow[8];
#pragma unroll
  for (int r = 0; r < 8; ++r) { mrow[r] = -1e30f; lrow[r] = 0.f; }
  v8f acc[4] = {};
  bf16_t* plds = pbuf[wave];

  const int NB = SEQ / 32;                             // 64 key blocks
  for (int i = 0; i < NB; ++i) {
    const int cur = i & 1;
    const int nb  = i * 32;
    if (wave == 0) __builtin_amdgcn_s_wait_tensorcnt((short)0);
    __syncthreads();                                   // cur tiles ready, other free
    if (wave == 0 && i + 1 < NB) {
      tdm_load_2d_bf16(lds_off(&ktile[1 - cur][0]), kbase + (size_t)(nb + 32) * HDIM,
                       HDIM, 32, HDIM, SEQ, HDIM);
      tdm_load_2d_bf16(lds_off(&vtile[1 - cur][0]), vbase + nb + 32,
                       32, HDIM, SEQ, HDIM, SEQ);
    }
    const bf16_t* kt = &ktile[cur][0];
    const bf16_t* vt = &vtile[cur][0];

    // S = (Q*scale) K^T for 32 keys: 4 K-frag loads, then 4 WMMAs
    v16bf kf0 = load_b_frag(kt + c * HDIM + 0  + boff);
    v16bf kf1 = load_b_frag(kt + c * HDIM + 32 + boff);
    v16bf kf2 = load_b_frag(kt + (16 + c) * HDIM + 0  + boff);
    v16bf kf3 = load_b_frag(kt + (16 + c) * HDIM + 32 + boff);
    v8f s0 = {}, s1 = {};
    s0 = wmma_bf16(qa0, kf0, s0);
    s0 = wmma_bf16(qa1, kf1, s0);
    s1 = wmma_bf16(qa0, kf2, s1);
    s1 = wmma_bf16(qa1, kf3, s1);

    // Issue V-frag LDS loads now; softmax VALU below hides their latency.
    v16bf vf[4];
#pragma unroll
    for (int t = 0; t < 4; ++t)
      vf[t] = load_b_frag(vt + (t * 16 + c) * 32 + boff);

    SCHED_GROUP(0x100, 8, 0);   // 8 K-frag DS reads
    SCHED_GROUP(0x008, 4, 0);   // 4 S WMMAs
    SCHED_GROUP(0x100, 8, 0);   // 8 V-frag DS reads (in flight over softmax)

    // online softmax; write P (bf16) into per-wave LDS slab
#pragma unroll
    for (int r = 0; r < 8; ++r) {
      const float a0 = s0[r], a1 = s1[r];
      float mx = half_max(fmaxf(a0, a1));
      const float nm   = fmaxf(mrow[r], mx);
      const float corr = __expf(mrow[r] - nm);
      const float p0   = __expf(a0 - nm);
      const float p1   = __expf(a1 - nm);
      const float rs   = half_sum(p0 + p1);
      lrow[r] = lrow[r] * corr + rs;
      mrow[r] = nm;
#pragma unroll
      for (int t = 0; t < 4; ++t) acc[t][r] *= corr;
      const int rowsel = r + ((lane >> 4) * 8);
      plds[rowsel * 32 + c]      = (bf16_t)p0;
      plds[rowsel * 32 + 16 + c] = (bf16_t)p1;
    }
    asm volatile("s_wait_dscnt 0" ::: "memory");       // cross-lane (intra-wave) LDS dep

    // O += P(16x32) * V(32x64); P is exactly one A-fragment from LDS
    const v16bf pa = load_a_frag(plds + c * 32 + aoff);
#pragma unroll
    for (int t = 0; t < 4; ++t)
      acc[t] = wmma_bf16(pa, vf[t], acc[t]);
  }

  const int b = bh >> 4, h = bh & 15;
#pragma unroll
  for (int r = 0; r < 8; ++r) {
    const float inv = 1.0f / lrow[r];
    const int s = row0 + r + ((lane >> 4) * 8);
    bf16_t* orow = ob + ((size_t)(b * SEQ + s)) * C_DIM + h * HDIM;
#pragma unroll
    for (int t = 0; t < 4; ++t) orow[t * 16 + c] = (bf16_t)(acc[t][r] * inv);
  }
}

// ---------------------------------------------------------------------------
// Kernel 4: output projection GEMM (M=4096, N=1024, K=1024) + bias -> fp32.
// ---------------------------------------------------------------------------
__global__ void __launch_bounds__(256) out_gemm(const bf16_t* __restrict__ ob,
                                                const bf16_t* __restrict__ wob,
                                                const float* __restrict__ b_out,
                                                float* __restrict__ out) {
  __shared__ __align__(16) bf16_t btile[2][64 * 32];

  const int lane = threadIdx.x & 31;
  const int wave = threadIdx.x >> 5;
  const int c    = lane & 15;
  const int aoff = (lane >> 4) * 8;
  const int boff = (lane >> 4) * 8;
  const int row_a = blockIdx.x * 128 + wave * 16 + c;
  const int col0  = blockIdx.y * 64;

  const bf16_t* arow = ob + (size_t)row_a * C_DIM + aoff;
  const bf16_t* bsrc = wob + (size_t)col0 * C_DIM;

  if (wave == 0)
    tdm_load_2d_bf16(lds_off(&btile[0][0]), bsrc, 32, 64, C_DIM, C_DIM, C_DIM);

  v8f acc[4] = {};
  v16bf aA = load_a_frag(arow);
  v16bf aB = aA;

  const int NI = C_DIM / 32;
  for (int j = 0; j < NI / 2 - 1; ++j) {
    const int i0 = 2 * j;
    if (wave == 0) __builtin_amdgcn_s_wait_tensorcnt((short)0);
    __syncthreads();
    if (wave == 0)
      tdm_load_2d_bf16(lds_off(&btile[1][0]), bsrc + (i0 + 1) * 32,
                       32, 64, C_DIM, C_DIM, C_DIM);
    aB = load_a_frag(arow + (i0 + 1) * 32);
    gemm_step(acc, aA, &btile[0][0], c, boff);
    if (wave == 0) __builtin_amdgcn_s_wait_tensorcnt((short)0);
    __syncthreads();
    if (wave == 0)
      tdm_load_2d_bf16(lds_off(&btile[0][0]), bsrc + (i0 + 2) * 32,
                       32, 64, C_DIM, C_DIM, C_DIM);
    aA = load_a_frag(arow + (i0 + 2) * 32);
    gemm_step(acc, aB, &btile[1][0], c, boff);
  }
  if (wave == 0) __builtin_amdgcn_s_wait_tensorcnt((short)0);
  __syncthreads();
  if (wave == 0)
    tdm_load_2d_bf16(lds_off(&btile[1][0]), bsrc + (NI - 1) * 32,
                     32, 64, C_DIM, C_DIM, C_DIM);
  aB = load_a_frag(arow + (NI - 1) * 32);
  gemm_step(acc, aA, &btile[0][0], c, boff);
  if (wave == 0) __builtin_amdgcn_s_wait_tensorcnt((short)0);
  __syncthreads();
  gemm_step(acc, aB, &btile[1][0], c, boff);

#pragma unroll
  for (int t = 0; t < 4; ++t) {
    const int o = col0 + t * 16 + c;
    const float bias = b_out[o];
#pragma unroll
    for (int r = 0; r < 8; ++r) {
      const int m = blockIdx.x * 128 + wave * 16 + ((lane >> 4) * 8) + r;
      out[(size_t)m * C_DIM + o] = acc[t][r] + bias;
    }
  }
}

// ---------------------------------------------------------------------------
extern "C" void kernel_launch(void* const* d_in, const int* in_sizes, int n_in,
                              void* d_out, int out_size, void* d_ws, size_t ws_size,
                              hipStream_t stream) {
  const float* x     = (const float*)d_in[0];
  const float* w_in  = (const float*)d_in[1];
  const float* b_in  = (const float*)d_in[2];
  const float* w_out = (const float*)d_in[3];
  const float* b_out = (const float*)d_in[4];
  float* out = (float*)d_out;

  char* ws = (char*)d_ws;
  size_t off = 0;
  auto alloc = [&](size_t bytes) -> void* {
    void* p = ws + off;
    off += (bytes + 255) & ~(size_t)255;
    return p;
  };
  bf16_t* xb  = (bf16_t*)alloc((size_t)MTOT * C_DIM * 2);
  bf16_t* wib = (bf16_t*)alloc((size_t)NQKV * C_DIM * 2);
  bf16_t* wob = (bf16_t*)alloc((size_t)C_DIM * C_DIM * 2);
  bf16_t* qb  = (bf16_t*)alloc((size_t)MTOT * C_DIM * 2);
  bf16_t* kb  = (bf16_t*)alloc((size_t)MTOT * C_DIM * 2);
  bf16_t* vtb = (bf16_t*)alloc((size_t)MTOT * C_DIM * 2);
  bf16_t* ob  = (bf16_t*)alloc((size_t)MTOT * C_DIM * 2);

  const int n_x = MTOT * C_DIM, n_wi = NQKV * C_DIM, n_wo = C_DIM * C_DIM;
  cvt_f32_bf16<<<(n_x  + 255) / 256, 256, 0, stream>>>(x,     xb,  n_x);
  cvt_f32_bf16<<<(n_wi + 255) / 256, 256, 0, stream>>>(w_in,  wib, n_wi);
  cvt_f32_bf16<<<(n_wo + 255) / 256, 256, 0, stream>>>(w_out, wob, n_wo);

  qkv_gemm<<<dim3(MTOT / 128, NQKV / 64), 256, 0, stream>>>(xb, wib, b_in, qb, kb, vtb);
  flash_attn<<<(BATCH * NHEAD * (SEQ / 16)) / 4, 128, 0, stream>>>(qb, kb, vtb, ob);
  out_gemm<<<dim3(MTOT / 128, C_DIM / 64), 256, 0, stream>>>(ob, wob, b_out, out);
}
